// CrossAttLayer_34316788695331
// MI455X (gfx1250) — matrix operand order repaired and checked by reading, not compiled, not run on hardware
//
#include <hip/hip_runtime.h>
#include <hip/hip_fp16.h>

// CDNA5 / gfx1250, wave32. WMMA shape 16x16x32 f16 -> f32.
typedef __attribute__((ext_vector_type(16))) _Float16 v16h;
typedef __attribute__((ext_vector_type(8)))  _Float16 v8h;
typedef __attribute__((ext_vector_type(8)))  float    v8f;
typedef __attribute__((ext_vector_type(4)))  int      v4i;

#define TT 91
#define FF 256
#define HH 4
#define DD 64
#define KFF 1024
#define AA 128
#define RR 1024

// ---- gfx1250 async load-to-LDS (ASYNCcnt) with portable fallback -----------
#if defined(__has_builtin)
#if __has_builtin(__builtin_amdgcn_global_load_async_to_lds_b128) && \
    __has_builtin(__builtin_amdgcn_s_wait_asynccnt)
#define USE_ASYNC_LDS 1
#endif
#endif
#ifndef USE_ASYNC_LDS
#define USE_ASYNC_LDS 0
#endif

static __device__ __forceinline__ void stage16B(const _Float16* g, _Float16* l) {
#if USE_ASYNC_LDS
  __builtin_amdgcn_global_load_async_to_lds_b128(
      (__attribute__((address_space(1))) v4i*)(unsigned long long)(uintptr_t)g,
      (__attribute__((address_space(3))) v4i*)(unsigned)(uintptr_t)l,
      0, 0);
#else
  *(v8h*)l = *(const v8h*)g;
#endif
}
static __device__ __forceinline__ void stage_wait() {
#if USE_ASYNC_LDS
  __builtin_amdgcn_s_wait_asynccnt(0);
#endif
}
// Wave-internal LDS RAW fence (LDS is in-order within a wave; this stops the
// compiler from reordering and waits out DScnt without a workgroup barrier).
static __device__ __forceinline__ void lds_wave_fence() {
  asm volatile("s_wait_dscnt 0" ::: "memory");
}

// A-matrix (16x32 f16) lane layout, ISA 7.12.2:
//   lane<16 : row M=lane,    K = {0..7, 16..23}
//   lane>=16: row M=lane-16, K = {8..15, 24..31}
// `row` points at X + m*ld + kb for this lane's row m; 16B-aligned runs.
static __device__ __forceinline__ v16h loadA_f32row(const float* __restrict__ row, int lane) {
  const int off = (lane & 16) ? 8 : 0;
  const float4 f0 = *(const float4*)(row + off);
  const float4 f1 = *(const float4*)(row + off + 4);
  const float4 f2 = *(const float4*)(row + off + 16);
  const float4 f3 = *(const float4*)(row + off + 20);
  v16h a;
  a[0] = (_Float16)f0.x; a[1] = (_Float16)f0.y; a[2] = (_Float16)f0.z; a[3] = (_Float16)f0.w;
  a[4] = (_Float16)f1.x; a[5] = (_Float16)f1.y; a[6] = (_Float16)f1.z; a[7] = (_Float16)f1.w;
  a[8]  = (_Float16)f2.x; a[9]  = (_Float16)f2.y; a[10] = (_Float16)f2.z; a[11] = (_Float16)f2.w;
  a[12] = (_Float16)f3.x; a[13] = (_Float16)f3.y; a[14] = (_Float16)f3.z; a[15] = (_Float16)f3.w;
  return a;
}

static __device__ __forceinline__ v16h loadA_f16row(const _Float16* __restrict__ row, int lane) {
  const int off = (lane & 16) ? 8 : 0;
  const v8h lo = *(const v8h*)(row + off);
  const v8h hi = *(const v8h*)(row + off + 16);
  return __builtin_shufflevector(lo, hi, 0, 1, 2, 3, 4, 5, 6, 7, 8, 9, 10, 11, 12, 13, 14, 15);
}

// ---------------------------------------------------------------------------
// Generic WMMA GEMM: epilogue(relu(X[M,K] @ W[K,N] + bias))
// block = 256 threads (8 waves); 128 rows x 64 cols per block; K chunks of 128.
// mode 0: scatter fp32+f16 [T][H][R][D]            (K/V proj; h = blockIdx.y)
// mode 1: scatter fp32+f16 [T][H][A][D] * q_scale  (Q proj)
// mode 2: outH[m][n] = f16(relu(.) + extra[m][n])  (S = relu(Y2) + agent)
// mode 3: outH[m][n] = f16(relu(.))                (F1)
// mode 4: outF[m][n] = relu(.)                     (F2, fp32 for LayerNorm)
// ---------------------------------------------------------------------------
template <typename XT>
__global__ __launch_bounds__(256) void gemm_kernel(
    const XT* __restrict__ X, const float* __restrict__ W,
    const float* __restrict__ bias, const float* __restrict__ extra,
    float* __restrict__ outF, _Float16* __restrict__ outH,
    int Kdim, int Nt, int mode)
{
  __shared__ _Float16 sWt[64][136];   // W panel transposed [n_local][k_local]

  const int tid  = threadIdx.x;
  const int lane = tid & 31;
  const int wave = tid >> 5;
  const int m0     = blockIdx.x * 128 + wave * 16;
  const int nPanel = blockIdx.y * 64;

  v8f acc[4] = {};
  const XT* xrow = X + (size_t)(m0 + (lane & 15)) * Kdim;

  for (int kc = 0; kc < Kdim; kc += 128) {
    // stage transposed f16 weight chunk [128 x 64]; float4 coalesced reads
    for (int idx = tid; idx < 16 * 128; idx += 256) {
      const int q = idx & 15, kl = idx >> 4;
      const float4 w4 = *(const float4*)&W[(size_t)(kc + kl) * Nt + nPanel + q * 4];
      sWt[q * 4 + 0][kl] = (_Float16)w4.x;
      sWt[q * 4 + 1][kl] = (_Float16)w4.y;
      sWt[q * 4 + 2][kl] = (_Float16)w4.z;
      sWt[q * 4 + 3][kl] = (_Float16)w4.w;
    }
    __syncthreads();
#pragma unroll
    for (int kk = 0; kk < 128; kk += 32) {
      if (kc + kk + 128 < Kdim) __builtin_prefetch(xrow + kc + kk + 128, 0, 1);
      v16h av;
      if constexpr (sizeof(XT) == 4) av = loadA_f32row((const float*)(xrow + kc + kk), lane);
      else                           av = loadA_f16row((const _Float16*)(xrow + kc + kk), lane);
      const int koff = kk + ((lane & 16) ? 8 : 0);
#pragma unroll
      for (int p = 0; p < 4; ++p) {
        const _Float16* bp = &sWt[p * 16 + (lane & 15)][koff];
        const v8h b0 = *(const v8h*)bp;
        const v8h b1 = *(const v8h*)(bp + 16);
        const v16h b = __builtin_shufflevector(b0, b1, 0, 1, 2, 3, 4, 5, 6, 7,
                                               8, 9, 10, 11, 12, 13, 14, 15);
        acc[p] = __builtin_amdgcn_wmma_f32_16x16x32_f16(
            false, av, false, b, (short)0, acc[p], false, false);
      }
    }
    __syncthreads();
  }

  // Epilogue. C layout: VGPR i, lane n -> M = i + (n<16?0:8), N = n&15.
#pragma unroll
  for (int p = 0; p < 4; ++p) {
#pragma unroll
    for (int i = 0; i < 8; ++i) {
      const int m = m0 + i + ((lane & 16) ? 8 : 0);
      const int n = nPanel + p * 16 + (lane & 15);
      float v = acc[p][i] + bias[n];
      v = fmaxf(v, 0.0f);
      if (mode == 0) {
        const int r = m / TT, t = m % TT;
        const size_t o = (((size_t)t * HH + blockIdx.y) * RR + r) * DD + (n & 63);
        outF[o] = v; outH[o] = (_Float16)v;
      } else if (mode == 1) {
        const int a_ = m / TT, t = m % TT;
        v *= extra[n & 63];   // q_scale (after ReLU, as in reference)
        const size_t o = (((size_t)t * HH + blockIdx.y) * AA + a_) * DD + (n & 63);
        outF[o] = v; outH[o] = (_Float16)v;
      } else if (mode == 2) {
        outH[(size_t)m * FF + n] = (_Float16)(v + extra[(size_t)m * FF + n]);
      } else if (mode == 3) {
        outH[(size_t)m * Nt + n] = (_Float16)v;
      } else {
        outF[(size_t)m * Nt + n] = v;
      }
    }
  }
}

// ---------------------------------------------------------------------------
// Fused attention for one (t,h): softmax(Q K^T / sqrt(H)) V, all f16 operands.
// block = 256 threads (8 waves); wave w owns agent rows w*16..w*16+15.
// 1/sqrt(H) folded into Q fragment. Pass 1: per-lane running maxima (cross-
// lane reduced once). Pass 2: exp + P@V with V staged transposed in LDS.
// ---------------------------------------------------------------------------
__global__ __launch_bounds__(256) void attn_kernel(
    const _Float16* __restrict__ Qh, const _Float16* __restrict__ Kh,
    const _Float16* __restrict__ Vh, _Float16* __restrict__ Y1h)
{
  __shared__ _Float16 sK[32 * 64];
  __shared__ _Float16 sVt[64][40];      // V tile transposed [d][r_local], padded
  __shared__ _Float16 sP[8][16 * 32];   // per-wave P tile (16 x 32)

  const int t = blockIdx.x, h = blockIdx.y;
  const int tid = threadIdx.x, lane = tid & 31, wave = tid >> 5;
  const _Float16* Qp = Qh + ((size_t)t * HH + h) * AA * DD;
  const _Float16* Kp = Kh + ((size_t)t * HH + h) * RR * DD;
  const _Float16* Vp = Vh + ((size_t)t * HH + h) * RR * DD;

  // Q in A-layout (x0.5 = 1/sqrt(H) folded in): two 16x32 k-steps over D=64
  const int arow = wave * 16 + (lane & 15);
  v16h qa[2];
#pragma unroll
  for (int ks = 0; ks < 2; ++ks) {
    qa[ks] = loadA_f16row(Qp + (size_t)arow * DD + ks * 32, lane);
    qa[ks] = qa[ks] * (_Float16)0.5f;
  }

  float rowmax[8], rowsum[8];
#pragma unroll
  for (int i = 0; i < 8; ++i) { rowmax[i] = -3.0e30f; rowsum[i] = 0.0f; }

  // ---- Pass 1: per-lane running maxima of energy ----
  for (int rb = 0; rb < RR; rb += 32) {
    __syncthreads();
    stage16B(Kp + (size_t)rb * 64 + tid * 8, sK + tid * 8);   // 4KB tile, 16B/thread
    stage_wait();
    __syncthreads();
#pragma unroll
    for (int sub = 0; sub < 32; sub += 16) {
      v8f e = {};
#pragma unroll
      for (int ks = 0; ks < 2; ++ks) {
        const _Float16* kp = &sK[(sub + (lane & 15)) * 64 + ks * 32 + ((lane & 16) ? 8 : 0)];
        const v16h b = __builtin_shufflevector(*(const v8h*)kp, *(const v8h*)(kp + 16), 0, 1, 2, 3, 4, 5, 6, 7, 8, 9, 10, 11, 12, 13, 14, 15);
        e = __builtin_amdgcn_wmma_f32_16x16x32_f16(false, qa[ks], false, b, (short)0, e, false, false);
      }
#pragma unroll
      for (int i = 0; i < 8; ++i) rowmax[i] = fmaxf(rowmax[i], e[i]);
    }
  }
  // one cross-lane max reduction per row (16-lane half-group)
#pragma unroll
  for (int i = 0; i < 8; ++i) {
    float ev = rowmax[i];
    ev = fmaxf(ev, __shfl_xor(ev, 1, 32));
    ev = fmaxf(ev, __shfl_xor(ev, 2, 32));
    ev = fmaxf(ev, __shfl_xor(ev, 4, 32));
    ev = fmaxf(ev, __shfl_xor(ev, 8, 32));
    rowmax[i] = ev;
  }

  v8f y1[4] = {};   // 16 x 64 output accumulator

  // ---- Pass 2: numerators + P @ V ----
  for (int rb = 0; rb < RR; rb += 32) {
    __syncthreads();
    stage16B(Kp + (size_t)rb * 64 + tid * 8, sK + tid * 8);
    {  // stage V transposed: one v8h read + 8 scatter stores per thread
      const int rl = tid >> 3, d0 = (tid & 7) * 8;
      const v8h v8 = *(const v8h*)(Vp + (size_t)(rb + rl) * 64 + d0);
#pragma unroll
      for (int c = 0; c < 8; ++c) sVt[d0 + c][rl] = v8[c];
    }
    stage_wait();
    __syncthreads();
#pragma unroll
    for (int sub = 0; sub < 32; sub += 16) {
      v8f e = {};
#pragma unroll
      for (int ks = 0; ks < 2; ++ks) {
        const _Float16* kp = &sK[(sub + (lane & 15)) * 64 + ks * 32 + ((lane & 16) ? 8 : 0)];
        const v16h b = __builtin_shufflevector(*(const v8h*)kp, *(const v8h*)(kp + 16), 0, 1, 2, 3, 4, 5, 6, 7, 8, 9, 10, 11, 12, 13, 14, 15);
        e = __builtin_amdgcn_wmma_f32_16x16x32_f16(false, qa[ks], false, b, (short)0, e, false, false);
      }
#pragma unroll
      for (int i = 0; i < 8; ++i) {
        const float pv = __expf(e[i] - rowmax[i]);
        rowsum[i] += pv;                      // per-lane partial; reduced after loop
        const int ml = i + ((lane & 16) ? 8 : 0);
        sP[wave][ml * 32 + sub + (lane & 15)] = (_Float16)pv;
      }
    }
    lds_wave_fence();   // sP is wave-private: in-order LDS + DScnt wait suffices
    {
      // P tile back as A-matrix (16 x 32 over local r)
      const v16h pa = loadA_f16row(&sP[wave][(lane & 15) * 32], lane);
      const int r0 = (lane & 16) ? 8 : 0;
#pragma unroll
      for (int p = 0; p < 4; ++p) {
        // B = V tile from transposed staging: two aligned b128 LDS loads
        const int d = p * 16 + (lane & 15);
        const v8h b0 = *(const v8h*)&sVt[d][r0];
        const v8h b1 = *(const v8h*)&sVt[d][r0 + 16];
        const v16h b = __builtin_shufflevector(b0, b1, 0, 1, 2, 3, 4, 5, 6, 7,
                                               8, 9, 10, 11, 12, 13, 14, 15);
        y1[p] = __builtin_amdgcn_wmma_f32_16x16x32_f16(false, pa, false, b, (short)0, y1[p], false, false);
      }
    }
  }

  // finish row sums (reduce across 16-lane half-group)
#pragma unroll
  for (int i = 0; i < 8; ++i) {
    rowsum[i] += __shfl_xor(rowsum[i], 1, 32);
    rowsum[i] += __shfl_xor(rowsum[i], 2, 32);
    rowsum[i] += __shfl_xor(rowsum[i], 4, 32);
    rowsum[i] += __shfl_xor(rowsum[i], 8, 32);
  }

#pragma unroll
  for (int p = 0; p < 4; ++p) {
#pragma unroll
    for (int i = 0; i < 8; ++i) {
      const int a_ = wave * 16 + i + ((lane & 16) ? 8 : 0);
      const int f  = h * DD + p * 16 + (lane & 15);
      Y1h[((size_t)a_ * TT + t) * FF + f] = (_Float16)(y1[p][i] / rowsum[i]);
    }
  }
}

// ---------------------------------------------------------------------------
// LayerNorm over last dim (256) of [A*T, 256]
// ---------------------------------------------------------------------------
__global__ __launch_bounds__(256) void ln_kernel(
    const float* __restrict__ X, const float* __restrict__ g,
    const float* __restrict__ b, float* __restrict__ Z)
{
  __shared__ float red[256];
  const int m = blockIdx.x, n = threadIdx.x;
  const float x = X[(size_t)m * 256 + n];
  red[n] = x;
  __syncthreads();
  for (int s = 128; s > 0; s >>= 1) {
    if (n < s) red[n] += red[n + s];
    __syncthreads();
  }
  const float mu = red[0] * (1.0f / 256.0f);
  __syncthreads();
  const float d = x - mu;
  red[n] = d * d;
  __syncthreads();
  for (int s = 128; s > 0; s >>= 1) {
    if (n < s) red[n] += red[n + s];
    __syncthreads();
  }
  const float var = red[0] * (1.0f / 256.0f);
  Z[(size_t)m * 256 + n] = d * rsqrtf(var + 1e-5f) * g[n] + b[n];
}

// ---------------------------------------------------------------------------
extern "C" void kernel_launch(void* const* d_in, const int* in_sizes, int n_in,
                              void* d_out, int out_size, void* d_ws, size_t ws_size,
                              hipStream_t stream) {
  (void)in_sizes; (void)n_in; (void)out_size; (void)ws_size;

  const float* agent   = (const float*)d_in[0];
  const float* rg      = (const float*)d_in[1];
  const float* W_K     = (const float*)d_in[2];
  const float* b_K     = (const float*)d_in[3];
  const float* W_V     = (const float*)d_in[4];
  const float* b_V     = (const float*)d_in[5];
  const float* W_Q     = (const float*)d_in[6];
  const float* b_Q     = (const float*)d_in[7];
  const float* q_scale = (const float*)d_in[8];
  const float* W_Y2    = (const float*)d_in[9];
  const float* b_Y2    = (const float*)d_in[10];
  const float* W_F1    = (const float*)d_in[11];
  const float* b_F1    = (const float*)d_in[12];
  const float* W_F2    = (const float*)d_in[13];
  const float* b_F2    = (const float*)d_in[14];
  const float* ln_g    = (const float*)d_in[15];
  const float* ln_b    = (const float*)d_in[16];

  // output tuple: (Z [A,T,F], Qh [T,H,A,D], Kh [T,H,R,D], Vh [T,H,R,D])
  float* Z  = (float*)d_out;
  float* Qh = Z  + (size_t)AA * TT * FF;
  float* Kh = Qh + (size_t)TT * HH * AA * DD;
  float* Vh = Kh + (size_t)TT * HH * RR * DD;

  // workspace (f16 copies of intermediates + fp32 F2 for LayerNorm)
  _Float16* Qh16 = (_Float16*)d_ws;
  _Float16* Kh16 = Qh16 + (size_t)TT * HH * AA * DD;
  _Float16* Vh16 = Kh16 + (size_t)TT * HH * RR * DD;
  _Float16* Y1h  = Vh16 + (size_t)TT * HH * RR * DD;
  _Float16* S16  = Y1h  + (size_t)AA * TT * FF;
  _Float16* F1h  = S16  + (size_t)AA * TT * FF;
  float*    F2f  = (float*)(F1h + (size_t)AA * TT * KFF);

  const dim3 blk(256);

  // K/V projections: rows = R*T = 93184 -> 728 row-blocks; 4 head panels
  gemm_kernel<float><<<dim3(728, 4), blk, 0, stream>>>(rg, W_K, b_K, nullptr, Kh, Kh16, FF, FF, 0);
  gemm_kernel<float><<<dim3(728, 4), blk, 0, stream>>>(rg, W_V, b_V, nullptr, Vh, Vh16, FF, FF, 0);
  // Q projection: rows = A*T = 11648 -> 91 row-blocks
  gemm_kernel<float><<<dim3(91, 4), blk, 0, stream>>>(agent, W_Q, b_Q, q_scale, Qh, Qh16, FF, FF, 1);
  // fused attention per (t, h)
  attn_kernel<<<dim3(TT, HH), blk, 0, stream>>>(Qh16, Kh16, Vh16, Y1h);
  // S = relu(Y1 @ W_Y2 + b_Y2) + agent
  gemm_kernel<_Float16><<<dim3(91, 4), blk, 0, stream>>>(Y1h, W_Y2, b_Y2, agent, nullptr, S16, FF, FF, 2);
  // FFN
  gemm_kernel<_Float16><<<dim3(91, 16), blk, 0, stream>>>(S16, W_F1, b_F1, nullptr, nullptr, F1h, FF, KFF, 3);
  gemm_kernel<_Float16><<<dim3(91, 4),  blk, 0, stream>>>(F1h, W_F2, b_F2, nullptr, F2f, nullptr, KFF, FF, 4);
  // LayerNorm -> Z
  ln_kernel<<<dim3(AA * TT), blk, 0, stream>>>(F2f, ln_g, ln_b, Z);
}
